// BiLSTM_CRF_17377437679865
// MI455X (gfx1250) — matrix operand order repaired and checked by reading, not compiled, not run on hardware
//
#include <hip/hip_runtime.h>
#include <hip/hip_bf16.h>

typedef __attribute__((ext_vector_type(16))) _Float16 v16h;
typedef __attribute__((ext_vector_type(8)))  _Float16 v8h;
typedef __attribute__((ext_vector_type(4)))  _Float16 v4h;
typedef __attribute__((ext_vector_type(8)))  float    v8f;

#define T_LEN 1024
#define B_SZ  128
#define E_SZ  256
#define G4H   512   // 4*H
#define H_SZ  128
#define NWG_PER_DIR 8

// ---------------- WMMA helpers (contiguous k order k = 16h + s, same for A and B) ----------------

__device__ __forceinline__ v8f wmma_f16(v16h a, v16h b, v8f c) {
    // (neg_a, A, neg_b, B, c_mod, C, reuse_a, reuse_b)
    return __builtin_amdgcn_wmma_f32_16x16x32_f16(false, a, false, b, (short)0, c,
                                                  false, false);
}

// A-matrix 16x32 f16 fragment: lane half h -> elems 0..7 = K(kk+16h+0..7),
// elems 8..15 = K(kk+16h+8+0..7). Two aligned 16B loads.
__device__ __forceinline__ v16h fragA_f16(const _Float16* row, int kk, int half) {
    v8h lo = *(const v8h*)(row + kk + half * 16);
    v8h hi = *(const v8h*)(row + kk + half * 16 + 8);
    v16h f;
#pragma unroll
    for (int i = 0; i < 8; ++i) { f[i] = lo[i]; f[8 + i] = hi[i]; }
    return f;
}

// B-matrix 32x16 f16 fragment: lane half h -> elems 0..15 = K(kk+16h+0..15):
// one aligned 32B load.
__device__ __forceinline__ v16h fragB_f16(const _Float16* p) {
    return *(const v16h*)p;
}

// ---------------- K0: zero the step counters ----------------

__global__ void k_init(int* cnt) {
    int i = blockIdx.x * 256 + threadIdx.x;
    if (i < 2 * T_LEN) cnt[i] = 0;
}

// ---------------- K0b: one-time f32 -> f16 conversion of weights/embeddings ----

__global__ void k_cvt(const float* __restrict__ src, _Float16* __restrict__ dst, int n) {
    int i = blockIdx.x * 256 + threadIdx.x;
    if (i < n) dst[i] = (_Float16)src[i];
}

// fc_w [15][256] padded to [16][256], row 15 = 0
__global__ void k_cvt_fcw(const float* __restrict__ src, _Float16* __restrict__ dst) {
    int i = blockIdx.x * 256 + threadIdx.x;
    if (i < 16 * 256) dst[i] = (i < 15 * 256) ? (_Float16)src[i] : (_Float16)0.f;
}

// ---------------- K1: embedding gather + input projection (both dirs) ----------------
// xproj layout: [t][n][b]  (b innermost) so the recurrence reads its C-init as
// one contiguous 8-half vector per lane (VGPR v <-> M=v+8*half <-> consecutive b).

__global__ void __launch_bounds__(256) k_embed_proj(
        const int* __restrict__ ids, const _Float16* __restrict__ emb16,
        const _Float16* __restrict__ wih16_f, const float* __restrict__ bias_f,
        const _Float16* __restrict__ wih16_b, const float* __restrict__ bias_b,
        _Float16* __restrict__ xp_f, _Float16* __restrict__ xp_b) {
    int wid  = blockIdx.x * 8 + (threadIdx.x >> 5);
    int lane = threadIdx.x & 31;
    int half = lane >> 4;
    int N    = lane & 15;
    int ng   = wid & 7;                // n-group: 4 tiles of 16 cols
    int rb   = (wid >> 3) & 8191;      // row block of 16 tokens (t-major flat index)
    int dir  = wid >> 16;

    const _Float16* wih  = dir ? wih16_b : wih16_f;
    const float*    bias = dir ? bias_b  : bias_f;
    _Float16*       xp   = dir ? xp_b    : xp_f;

    int rr = rb * 16 + N;              // flat token row for this lane (A matrix M)
    int t  = rr >> 7;
    int b  = rr & 127;
    int id = ids[b * T_LEN + t];       // emb row 0 is already zero (padding_idx)
    const _Float16* arow = emb16 + (size_t)id * E_SZ;

    int n0 = ng * 64;
    v8f acc[4];
#pragma unroll
    for (int j = 0; j < 4; ++j) { v8f z = {}; acc[j] = z; }

    for (int kk = 0; kk < E_SZ; kk += 32) {
        v16h a = fragA_f16(arow, kk, half);
#pragma unroll
        for (int j = 0; j < 4; ++j) {
            v16h bb = fragB_f16(wih + (size_t)(n0 + j * 16 + N) * E_SZ + kk + half * 16);
            acc[j] = wmma_f16(a, bb, acc[j]);
        }
    }
    int tt    = (rb * 16) >> 7;        // tile spans a single t
    int bbase = ((rb * 16) & 127) + 8 * half;
#pragma unroll
    for (int j = 0; j < 4; ++j) {
        int n = n0 + j * 16 + N;       // all elements of this lane share N
        float bj = bias[n];
        v8h pk;
#pragma unroll
        for (int v = 0; v < 8; ++v) pk[v] = (_Float16)(acc[j][v] + bj);
        *(v8h*)(xp + ((size_t)tt * G4H + n) * B_SZ + bbase) = pk;   // 16B store
    }
}

// ---------------- K2: bidirectional LSTM recurrence ----------------
// 16 workgroups: blocks 0..7 forward (h cols 16j..), 8..15 backward.
// Per step: gates = xproj_t + h_prev @ w_hh^T (WMMA), nonlinearity, write h
// slice to h_all, global atomic-counter barrier across the 8 WGs of a direction.

__global__ void __launch_bounds__(512) k_lstm(
        const _Float16* __restrict__ whh16_f, const _Float16* __restrict__ whh16_b,
        const _Float16* __restrict__ xp_f, const _Float16* __restrict__ xp_b,
        _Float16* __restrict__ h_all, int* __restrict__ cnt_f, int* __restrict__ cnt_b) {
    __shared__ __align__(32) _Float16 wsl[64 * H_SZ];   // 16 KB: w_hh slice (f16)
    __shared__ float gbuf[4 * B_SZ * 16];               // 32 KB: gate staging (f32)

    int dir  = blockIdx.x >> 3;
    int hcol = blockIdx.x & 7;                          // h columns 16*hcol .. +15
    const _Float16* whh = dir ? whh16_b : whh16_f;
    const _Float16* xp  = dir ? xp_b    : xp_f;
    int*            cnt = dir ? cnt_b   : cnt_f;

    int tid = threadIdx.x;

    // stage the 64 w_hh rows (4 gates x 16 cols) this WG needs
    for (int e = tid; e < 64 * H_SZ; e += 512) {
        int rowl = e >> 7;                  // g*16 + c
        int k    = e & 127;
        int g = rowl >> 4, c = rowl & 15;
        int n = g * H_SZ + hcol * 16 + c;   // row of w_hh [512 x 128]
        wsl[e] = whh[(size_t)n * H_SZ + k];
    }
    __syncthreads();

    int lane = tid & 31, half = lane >> 4, N = lane & 15;
    int w = tid >> 5;
    int r = w & 7;                          // row block (16 batch rows)
    int p = w >> 3;                         // gate pair: gates 2p, 2p+1
    int bbase = r * 16 + 8 * half;

    float cst[4] = {0.f, 0.f, 0.f, 0.f};    // persistent cell state (f32, in regs)
    int eb[4], ec[4];
#pragma unroll
    for (int e = 0; e < 4; ++e) { int idx = tid * 4 + e; eb[e] = idx >> 4; ec[e] = idx & 15; }

    for (int stepi = 0; stepi < T_LEN; ++stepi) {
        int t = dir ? (T_LEN - 1 - stepi) : stepi;
        v8f acc0 = {}, acc1 = {};

        // prefetch next step's xproj lines into L2/near caches
        if (stepi + 1 < T_LEN) {
            int tn = dir ? (t - 1) : (t + 1);
#pragma unroll
            for (int i = 0; i < 2; ++i) {
                int n = (2 * p + i) * H_SZ + hcol * 16 + N;
                __builtin_prefetch(xp + ((size_t)tn * G4H + n) * B_SZ + bbase, 0, 1);
            }
        }

        if (stepi != 0) {
            int tprev = dir ? (t + 1) : (t - 1);
            const _Float16* arow =
                h_all + ((size_t)(tprev * B_SZ) + r * 16 + N) * 256 + dir * H_SZ;
#pragma unroll
            for (int kk = 0; kk < H_SZ; kk += 32) {
                v16h a  = fragA_f16(arow, kk, half);
                v16h b0 = fragB_f16(wsl + ((2 * p) * 16 + N) * H_SZ + kk + half * 16);
                acc0 = wmma_f16(a, b0, acc0);
                v16h b1 = fragB_f16(wsl + ((2 * p + 1) * 16 + N) * H_SZ + kk + half * 16);
                acc1 = wmma_f16(a, b1, acc1);
            }
        }
        // add input projection (one contiguous 16B vector per tile), stage gates
#pragma unroll
        for (int i = 0; i < 2; ++i) {
            int g = 2 * p + i;
            int n = g * H_SZ + hcol * 16 + N;
            v8h xv = *(const v8h*)(xp + ((size_t)t * G4H + n) * B_SZ + bbase);
            v8f acc = i ? acc1 : acc0;
#pragma unroll
            for (int v = 0; v < 8; ++v) {
                int brow = r * 16 + v + 8 * half;
                gbuf[g * 2048 + brow * 16 + N] = acc[v] + (float)xv[v];
            }
        }
        __syncthreads();

        // elementwise gate math: 4 (b,c) elements per thread, packed 8B h store
        v4h hpk;
#pragma unroll
        for (int e = 0; e < 4; ++e) {
            int o = eb[e] * 16 + ec[e];
            float gi = gbuf[0 * 2048 + o];
            float gf = gbuf[1 * 2048 + o];
            float gg = gbuf[2 * 2048 + o];
            float go = gbuf[3 * 2048 + o];
            float si = 1.f / (1.f + __expf(-gi));
            float sf = 1.f / (1.f + __expf(-gf));
            float so = 1.f / (1.f + __expf(-go));
            cst[e] = sf * cst[e] + si * tanhf(gg);
            hpk[e] = (_Float16)(so * tanhf(cst[e]));
        }
        *(v4h*)(h_all + ((size_t)t * B_SZ + eb[0]) * 256 + dir * H_SZ + hcol * 16 + ec[0]) =
            hpk;
        __syncthreads();

        // cross-WGP step barrier (8 WGs per direction, co-resident)
        if (tid == 0) {
            __threadfence();                      // release h stores to device scope
            atomicAdd(cnt + stepi, 1);
            while (__hip_atomic_load(cnt + stepi, __ATOMIC_RELAXED,
                                     __HIP_MEMORY_SCOPE_AGENT) < NWG_PER_DIR)
                __builtin_amdgcn_s_sleep(1);
        }
        __syncthreads();
        __threadfence();                          // acquire: invalidate near caches
    }
}

// ---------------- K3: emissions = h_all @ fc_w^T + fc_b  (L padded to 16) ----------------

__global__ void __launch_bounds__(256) k_emis(
        const _Float16* __restrict__ h_all, const _Float16* __restrict__ fcw16,
        const float* __restrict__ fcb, float* __restrict__ em) {
    int rb   = blockIdx.x * 8 + (threadIdx.x >> 5);   // 0..8191 row tiles
    int lane = threadIdx.x & 31;
    int half = lane >> 4, N = lane & 15;

    const _Float16* arow = h_all + ((size_t)rb * 16 + N) * 256;
    const _Float16* brow = fcw16 + (size_t)N * E_SZ;  // row 15 is zero-padded

    v8f acc = {};
    for (int kk = 0; kk < E_SZ; kk += 32) {
        v16h a  = fragA_f16(arow, kk, half);
        v16h bb = fragB_f16(brow + kk + half * 16);
        acc = wmma_f16(a, bb, acc);
    }
    float bias = (N < 15) ? fcb[N] : 0.f;
#pragma unroll
    for (int v = 0; v < 8; ++v) {
        int row = rb * 16 + v + 8 * half;             // row = t*128 + b
        em[(size_t)row * 16 + N] = acc[v] + bias;
    }
}

// ---------------- K4: CRF numerator + partition, one wave per batch row ----------------

__global__ void __launch_bounds__(128) k_crf(
        const float* __restrict__ em, const int* __restrict__ labels,
        const unsigned char* __restrict__ mask,
        const float* __restrict__ start_t, const float* __restrict__ end_t,
        const float* __restrict__ trans, float* __restrict__ res) {
    int b    = blockIdx.x * 4 + (threadIdx.x >> 5);
    int lane = threadIdx.x & 31;

    // ---- numerator: lane-parallel over time ----
    float part = 0.f; int mcount = 0;
    for (int t = lane; t < T_LEN; t += 32) {
        int mt = mask[b * T_LEN + t] ? 1 : 0;
        mcount += mt;
        if (t > 0 && mt) {
            int tp = labels[b * T_LEN + t - 1];
            int tc = labels[b * T_LEN + t];
            part += trans[tp * 15 + tc] + em[((size_t)t * B_SZ + b) * 16 + tc];
        }
    }
#pragma unroll
    for (int off = 16; off; off >>= 1) {
        part   += __shfl_down(part, off, 32);
        mcount += __shfl_down(mcount, off, 32);
    }
    part   = __shfl(part, 0, 32);
    mcount = __shfl(mcount, 0, 32);
    int tag0 = labels[b * T_LEN];
    float score = part + start_t[tag0] + em[(size_t)b * 16 + tag0];
    int last_tag = labels[b * T_LEN + (mcount - 1)];
    score += end_t[last_tag];

    // ---- partition (forward algorithm): lane j holds alpha_j, j<15 ----
    int j = (lane < 15) ? lane : 14;
    float tcol[15];
#pragma unroll
    for (int i = 0; i < 15; ++i) tcol[i] = trans[i * 15 + j];
    float alpha = start_t[j] + em[(size_t)b * 16 + j];

    for (int t = 1; t < T_LEN; ++t) {
        float emj = em[((size_t)t * B_SZ + b) * 16 + j];
        int mt = mask[b * T_LEN + t];
        float vbuf[15];
        float m = -1e30f;
#pragma unroll
        for (int i = 0; i < 15; ++i) {
            float ai = __shfl(alpha, i, 32);
            vbuf[i] = ai + tcol[i];
            m = fmaxf(m, vbuf[i]);
        }
        float s = 0.f;
#pragma unroll
        for (int i = 0; i < 15; ++i) s += __expf(vbuf[i] - m);
        float nxt = m + __logf(s) + emj;
        if (mt) alpha = nxt;
    }
    float vj = (lane < 15) ? (alpha + end_t[j]) : -1e30f;
    float mm = vj;
#pragma unroll
    for (int off = 16; off; off >>= 1) mm = fmaxf(mm, __shfl_xor(mm, off, 32));
    float ss = (lane < 15) ? __expf(vj - mm) : 0.f;
#pragma unroll
    for (int off = 16; off; off >>= 1) ss += __shfl_xor(ss, off, 32);
    float denom = mm + __logf(ss);

    if (lane == 0) res[b] = score - denom;
}

// ---------------- K5: deterministic final mean ----------------

__global__ void k_final(const float* __restrict__ res, float* __restrict__ out) {
    int lane = threadIdx.x;
    float s = res[lane] + res[lane + 32] + res[lane + 64] + res[lane + 96];
#pragma unroll
    for (int off = 16; off; off >>= 1) s += __shfl_xor(s, off, 32);
    if (lane == 0) out[0] = -s / 128.f;
}

// ---------------- host-side launch ----------------

extern "C" void kernel_launch(void* const* d_in, const int* in_sizes, int n_in,
                              void* d_out, int out_size, void* d_ws, size_t ws_size,
                              hipStream_t stream) {
    const int*           ids    = (const int*)d_in[0];
    const unsigned char* mask   = (const unsigned char*)d_in[1];
    const int*           labels = (const int*)d_in[2];
    const float*         emb    = (const float*)d_in[3];
    const float*         wih_f  = (const float*)d_in[4];
    const float*         whh_f  = (const float*)d_in[5];
    const float*         b_f    = (const float*)d_in[6];
    const float*         wih_b  = (const float*)d_in[7];
    const float*         whh_b  = (const float*)d_in[8];
    const float*         b_b    = (const float*)d_in[9];
    const float*         fcw    = (const float*)d_in[10];
    const float*         fcb    = (const float*)d_in[11];
    const float*         st     = (const float*)d_in[12];
    const float*         et     = (const float*)d_in[13];
    const float*         tr     = (const float*)d_in[14];

    char* ws = (char*)d_ws;
    const size_t SZ_XP   = (size_t)T_LEN * B_SZ * G4H * sizeof(_Float16); // 128 MiB
    const size_t SZ_HALL = (size_t)T_LEN * B_SZ * 256 * sizeof(_Float16); //  64 MiB
    const size_t SZ_EM   = (size_t)T_LEN * B_SZ * 16 * sizeof(float);     //   8 MiB
    const size_t O_MISC  = 2 * SZ_XP + SZ_HALL + SZ_EM;

    _Float16* xp_f    = (_Float16*)(ws);
    _Float16* xp_b    = (_Float16*)(ws + SZ_XP);
    _Float16* h_all   = (_Float16*)(ws + 2 * SZ_XP);
    float*    em      = (float*)(ws + 2 * SZ_XP + SZ_HALL);
    int*      cnt     = (int*)(ws + O_MISC);                     //  8 KiB
    float*    res     = (float*)(ws + O_MISC + 8192);            // 512 B
    _Float16* emb16   = (_Float16*)(ws + O_MISC + 16384);        // 128 KiB
    _Float16* wih16_f = (_Float16*)(ws + O_MISC + 147456);       // 256 KiB
    _Float16* wih16_b = (_Float16*)(ws + O_MISC + 409600);       // 256 KiB
    _Float16* whh16_f = (_Float16*)(ws + O_MISC + 671744);       // 128 KiB
    _Float16* whh16_b = (_Float16*)(ws + O_MISC + 802816);       // 128 KiB
    _Float16* fcw16   = (_Float16*)(ws + O_MISC + 933888);       //   8 KiB
    float*    out     = (float*)d_out;

    k_init<<<8, 256, 0, stream>>>(cnt);
    k_cvt<<<(256 * 256 + 255) / 256, 256, 0, stream>>>(emb, emb16, 256 * 256);
    k_cvt<<<(512 * 256 + 255) / 256, 256, 0, stream>>>(wih_f, wih16_f, 512 * 256);
    k_cvt<<<(512 * 256 + 255) / 256, 256, 0, stream>>>(wih_b, wih16_b, 512 * 256);
    k_cvt<<<(512 * 128 + 255) / 256, 256, 0, stream>>>(whh_f, whh16_f, 512 * 128);
    k_cvt<<<(512 * 128 + 255) / 256, 256, 0, stream>>>(whh_b, whh16_b, 512 * 128);
    k_cvt_fcw<<<16, 256, 0, stream>>>(fcw, fcw16);

    k_embed_proj<<<16384, 256, 0, stream>>>(ids, emb16, wih16_f, b_f, wih16_b, b_b,
                                            xp_f, xp_b);
    k_lstm<<<16, 512, 0, stream>>>(whh16_f, whh16_b, xp_f, xp_b, h_all,
                                   cnt, cnt + T_LEN);
    k_emis<<<1024, 256, 0, stream>>>(h_all, fcw16, fcb, em);
    k_crf<<<32, 128, 0, stream>>>(em, labels, mask, st, et, tr, res);
    k_final<<<1, 32, 0, stream>>>(res, out);
}
